// XCA_37752762532413
// MI455X (gfx1250) — matrix operand order repaired and checked by reading, not compiled, not run on hardware
//
#include <hip/hip_runtime.h>
#include <hip/hip_bf16.h>
#include <math.h>

// ---------------- types ----------------
typedef __attribute__((ext_vector_type(16))) __bf16 v16bf;
typedef __attribute__((ext_vector_type(8)))  __bf16 v8bf;
typedef __attribute__((ext_vector_type(8)))  float  v8f;

#define DEVINL __device__ __forceinline__

static constexpr int Bc = 2, Nc = 2048, Cc = 1024, Hc = 16, Dc = 64;
static constexpr int Mrows = Bc * Nc;          // 4096
static constexpr int QKVc  = 3 * Cc;           // 3072

DEVINL v8f vzero8() {
  v8f r;
  #pragma unroll
  for (int i = 0; i < 8; ++i) r[i] = 0.0f;
  return r;
}

DEVINL v16bf cat8(v8bf lo, v8bf hi) {
  return __builtin_shufflevector(lo, hi, 0,1,2,3,4,5,6,7,8,9,10,11,12,13,14,15);
}

DEVINL v8f wmma_bf16(v16bf a, v16bf b, v8f c) {
  return __builtin_amdgcn_wmma_f32_16x16x32_bf16(false, a, false, b, (short)0, c,
                                                 false, false);
}

// ---------------- kernel 0: one-time fp32 -> bf16 conversion ---------------
// n must be a multiple of 8 (true for all buffers here).
__global__ void __launch_bounds__(256)
cvt_bf16_kernel(const float* __restrict__ src, __bf16* __restrict__ dst,
                int n8 /* = n/8 */) {
  const int i = blockIdx.x * 256 + threadIdx.x;
  if (i >= n8) return;
  const v8f v = *(const v8f*)(src + (size_t)i * 8);
  v8bf o;
  #pragma unroll
  for (int j = 0; j < 8; ++j) o[j] = (__bf16)v[j];
  *(v8bf*)(dst + (size_t)i * 8) = o;
}

// ---------------- kernel 1: qkv = x @ qkv_w^T  (bf16 in -> fp32 out) -------
// A:[4096,1024] bf16 row-major, W:[3072,1024] bf16 row-major, Y:[4096,3072]
// block 256 = 8 waves; each wave computes a 16x64 output strip.
__global__ void __launch_bounds__(256)
gemm_qkv_kernel(const __bf16* __restrict__ A, const __bf16* __restrict__ W,
                float* __restrict__ Y) {
  const int lane = threadIdx.x & 31;
  const int wave = threadIdx.x >> 5;
  const int l16  = lane & 15;
  const int g    = lane >> 4;            // 0 or 1 (lane half)
  const int kb   = g * 16;               // A fragment sub-chunk offset
  const int m0   = (blockIdx.y * 8 + wave) * 16;
  const int n0   = blockIdx.x * 64;

  v8f acc[4] = {};
  const __bf16* arow = A + (size_t)(m0 + l16) * Cc;

  for (int k0 = 0; k0 < Cc; k0 += 32) {
    v16bf afrag = cat8(*(const v8bf*)(arow + k0 + kb),
                       *(const v8bf*)(arow + k0 + 8 + kb));
    #pragma unroll
    for (int t = 0; t < 4; ++t) {
      const __bf16* brow = W + (size_t)(n0 + t * 16 + l16) * Cc + k0 + g * 16;
      acc[t] = wmma_bf16(afrag, *(const v16bf*)brow, acc[t]);
    }
  }
  #pragma unroll
  for (int t = 0; t < 4; ++t)
    #pragma unroll
    for (int r = 0; r < 8; ++r)
      Y[(size_t)(m0 + r + 8 * g) * QKVc + n0 + t * 16 + l16] = acc[t][r];
}

// ------------- kernel 2: L2-normalize q,k over C; split heads; bf16 --------
// qkv:[4096,3072] fp32. Outputs:
//   qb: [B,H,N,D] bf16, scaled by 1/sqrt(D) * 1/max(||.||,eps) * 1/temp[h]
//   kb: [B,H,N,D] bf16, scaled by 1/sqrt(D) * 1/max(||.||,eps)
//   vt: [B,H,D,N] bf16 (transposed so attention B-fragments are contiguous)
__global__ void __launch_bounds__(256)
norm_split_kernel(const float* __restrict__ QKV,
                  const float* __restrict__ temperature,
                  __bf16* __restrict__ qb, __bf16* __restrict__ kb,
                  __bf16* __restrict__ vt) {
  __shared__ float rq[256];
  __shared__ float rk[256];
  const int row = blockIdx.x;            // 0..4095
  const int tid = threadIdx.x;
  const float* src = QKV + (size_t)row * QKVc;

  float sq = 0.f, sk = 0.f;
  for (int c = tid; c < Cc; c += 256) {
    float a = src[c];        sq += a * a;
    float b = src[c + Cc];   sk += b * b;
  }
  rq[tid] = sq; rk[tid] = sk;
  __syncthreads();
  for (int s = 128; s > 0; s >>= 1) {
    if (tid < s) { rq[tid] += rq[tid + s]; rk[tid] += rk[tid + s]; }
    __syncthreads();
  }
  const float scale = 0.125f;            // 1/sqrt(D)
  const float fq = scale / fmaxf(sqrtf(rq[0]), 1e-12f);
  const float fk = scale / fmaxf(sqrtf(rk[0]), 1e-12f);

  const int b = row >> 11;               // /N
  const int n = row & (Nc - 1);
  for (int c = tid; c < Cc; c += 256) {
    const int h = c >> 6, d = c & 63;
    const size_t bh = (size_t)(b * Hc + h);
    const float fqh = fq / temperature[h];   // fold 1/temp into q (exact)
    qb[(bh * Nc + n) * Dc + d] = (__bf16)(src[c] * fqh);
    kb[(bh * Nc + n) * Dc + d] = (__bf16)(src[c + Cc] * fk);
    vt[(bh * Dc + d) * Nc + n] = (__bf16)(src[c + 2 * Cc]);
  }
}

// ------------- kernel 3: flash attention per (b,h), 16-row q tile ----------
// grid = (32 tile-groups, 32 bh), block 128 (4 waves, one q-tile each).
// KV processed in blocks of 32 keys: 4 WMMA for S (16x32), streaming softmax,
// P relayout via per-wave LDS, 4 WMMA for P@V into 16x64 accumulators.
__global__ void __launch_bounds__(128)
attn_kernel(const __bf16* __restrict__ qb, const __bf16* __restrict__ kbuf,
            const __bf16* __restrict__ vt, __bf16* __restrict__ ob) {
  __shared__ __align__(64) __bf16 pshared[4][16 * 32];

  const int lane = threadIdx.x & 31;
  const int wave = threadIdx.x >> 5;
  const int l16  = lane & 15;
  const int g    = lane >> 4;
  const int kbA  = g * 16;
  const int bh   = blockIdx.y;
  const int h    = bh & (Hc - 1);
  const int b    = bh >> 4;
  const int q0   = (blockIdx.x * 4 + wave) * 16;

  __bf16* pw = pshared[wave];

  // q A-fragments for the two 32-wide d-halves
  const __bf16* qrow = qb + ((size_t)bh * Nc + q0 + l16) * Dc;
  v16bf qA[2];
  #pragma unroll
  for (int s = 0; s < 2; ++s)
    qA[s] = cat8(*(const v8bf*)(qrow + s * 32 + kbA),
                 *(const v8bf*)(qrow + s * 32 + 8 + kbA));

  float mrow[8], lrow[8];
  #pragma unroll
  for (int r = 0; r < 8; ++r) { mrow[r] = -INFINITY; lrow[r] = 0.0f; }
  v8f acc[4] = {};

  for (int kv0 = 0; kv0 < Nc; kv0 += 32) {
    // ---- S = q @ k^T for 32 keys (two 16x16 tiles) ----
    v8f S0 = vzero8(), S1 = vzero8();
    {
      const __bf16* kr0 = kbuf + ((size_t)bh * Nc + kv0 + l16) * Dc + g * 16;
      const __bf16* kr1 = kr0 + 16 * Dc;
      S0 = wmma_bf16(qA[0], *(const v16bf*)kr0,        S0);
      S0 = wmma_bf16(qA[1], *(const v16bf*)(kr0 + 32), S0);
      S1 = wmma_bf16(qA[0], *(const v16bf*)kr1,        S1);
      S1 = wmma_bf16(qA[1], *(const v16bf*)(kr1 + 32), S1);
    }

    // ---- streaming softmax: row stats live replicated in each lane-half ----
    float mloc[8];
    #pragma unroll
    for (int r = 0; r < 8; ++r) mloc[r] = fmaxf(S0[r], S1[r]);
    #pragma unroll
    for (int off = 8; off > 0; off >>= 1)
      #pragma unroll
      for (int r = 0; r < 8; ++r)
        mloc[r] = fmaxf(mloc[r], __shfl_xor(mloc[r], off, 32));

    float alpha[8];
    #pragma unroll
    for (int r = 0; r < 8; ++r) {
      const float mn = fmaxf(mrow[r], mloc[r]);
      alpha[r] = __expf(mrow[r] - mn);
      mrow[r]  = mn;
    }
    #pragma unroll
    for (int i = 0; i < 8; ++i) {
      S0[i] = __expf(S0[i] - mrow[i]);
      S1[i] = __expf(S1[i] - mrow[i]);
    }
    float ls[8];
    #pragma unroll
    for (int r = 0; r < 8; ++r) ls[r] = S0[r] + S1[r];
    #pragma unroll
    for (int off = 8; off > 0; off >>= 1)
      #pragma unroll
      for (int r = 0; r < 8; ++r) ls[r] += __shfl_xor(ls[r], off, 32);
    #pragma unroll
    for (int r = 0; r < 8; ++r) lrow[r] = lrow[r] * alpha[r] + ls[r];

    // ---- relayout P (C/D layout -> A fragment) through per-wave LDS ----
    #pragma unroll
    for (int r = 0; r < 8; ++r) {
      const int m = r + 8 * g;
      pw[m * 32 + l16]      = (__bf16)S0[r];
      pw[m * 32 + 16 + l16] = (__bf16)S1[r];
    }
    // same-wave DS ops are in order; keep the compiler from reordering
    asm volatile("" ::: "memory");
    v16bf pA = cat8(*(const v8bf*)(pw + l16 * 32 + kbA),
                    *(const v8bf*)(pw + l16 * 32 + 8 + kbA));

    // ---- out += P @ V (V pre-transposed -> contiguous B fragments) ----
    #pragma unroll
    for (int t = 0; t < 4; ++t) {
      const __bf16* vr =
          vt + ((size_t)bh * Dc + t * 16 + l16) * Nc + kv0 + g * 16;
      #pragma unroll
      for (int i = 0; i < 8; ++i) acc[t][i] *= alpha[i];
      acc[t] = wmma_bf16(pA, *(const v16bf*)vr, acc[t]);
    }
  }

  // epilogue: normalize by row sum, store as [B,N,C] bf16 for the proj GEMM
  #pragma unroll
  for (int t = 0; t < 4; ++t)
    #pragma unroll
    for (int r = 0; r < 8; ++r) {
      const float o = acc[t][r] / lrow[r];
      ob[((size_t)b * Nc + q0 + r + 8 * g) * Cc + h * Dc + t * 16 + l16] =
          (__bf16)o;
    }
}

// ------------- kernel 4: out = attn_out @ proj_w^T + bias (fp32 out) -------
__global__ void __launch_bounds__(256)
gemm_proj_kernel(const __bf16* __restrict__ A, const __bf16* __restrict__ W,
                 const float* __restrict__ bias, float* __restrict__ Y) {
  const int lane = threadIdx.x & 31;
  const int wave = threadIdx.x >> 5;
  const int l16  = lane & 15;
  const int g    = lane >> 4;
  const int kb   = g * 16;
  const int m0   = (blockIdx.y * 8 + wave) * 16;
  const int n0   = blockIdx.x * 64;

  v8f acc[4] = {};
  const __bf16* arow = A + (size_t)(m0 + l16) * Cc;

  for (int k0 = 0; k0 < Cc; k0 += 32) {
    v16bf afrag = cat8(*(const v8bf*)(arow + k0 + kb),
                       *(const v8bf*)(arow + k0 + 8 + kb));
    #pragma unroll
    for (int t = 0; t < 4; ++t) {
      const __bf16* brow = W + (size_t)(n0 + t * 16 + l16) * Cc + k0 + g * 16;
      acc[t] = wmma_bf16(afrag, *(const v16bf*)brow, acc[t]);
    }
  }
  #pragma unroll
  for (int t = 0; t < 4; ++t) {
    const int n = n0 + t * 16 + l16;
    const float bv = bias[n];
    #pragma unroll
    for (int r = 0; r < 8; ++r)
      Y[(size_t)(m0 + r + 8 * g) * Cc + n] = acc[t][r] + bv;
  }
}

// ---------------------------- launcher -------------------------------------
extern "C" void kernel_launch(void* const* d_in, const int* in_sizes, int n_in,
                              void* d_out, int out_size, void* d_ws,
                              size_t ws_size, hipStream_t stream) {
  (void)in_sizes; (void)n_in; (void)out_size; (void)ws_size;

  const float* x           = (const float*)d_in[0];  // [B,N,C]
  const float* qkv_w       = (const float*)d_in[1];  // [3C,C]
  const float* temperature = (const float*)d_in[2];  // [H]
  const float* proj_w      = (const float*)d_in[3];  // [C,C]
  const float* proj_b      = (const float*)d_in[4];  // [C]
  float* out               = (float*)d_out;          // [B,N,C]

  char* ws = (char*)d_ws;
  size_t off = 0;
  float* qkv32  = (float*)(ws + off);  off += (size_t)Mrows * QKVc * sizeof(float); // 48 MB
  __bf16* qb    = (__bf16*)(ws + off); off += (size_t)Bc * Hc * Nc * Dc * 2;        //  8 MB
  __bf16* kb    = (__bf16*)(ws + off); off += (size_t)Bc * Hc * Nc * Dc * 2;        //  8 MB
  __bf16* vt    = (__bf16*)(ws + off); off += (size_t)Bc * Hc * Nc * Dc * 2;        //  8 MB
  __bf16* ob    = (__bf16*)(ws + off); off += (size_t)Mrows * Cc * 2;               //  8 MB
  __bf16* xb    = (__bf16*)(ws + off); off += (size_t)Mrows * Cc * 2;               //  8 MB
  __bf16* qkvwb = (__bf16*)(ws + off); off += (size_t)QKVc * Cc * 2;                //  6 MB
  __bf16* pwb   = (__bf16*)(ws + off); off += (size_t)Cc * Cc * 2;                  //  2 MB

  // 0) one-time bf16 conversion of GEMM operands (memory-bound, ~1 us total)
  {
    const int n8x = Mrows * Cc / 8, n8w = QKVc * Cc / 8, n8p = Cc * Cc / 8;
    cvt_bf16_kernel<<<(n8x + 255) / 256, 256, 0, stream>>>(x, xb, n8x);
    cvt_bf16_kernel<<<(n8w + 255) / 256, 256, 0, stream>>>(qkv_w, qkvwb, n8w);
    cvt_bf16_kernel<<<(n8p + 255) / 256, 256, 0, stream>>>(proj_w, pwb, n8p);
  }
  // 1) qkv = x @ qkv_w^T
  gemm_qkv_kernel<<<dim3(QKVc / 64, Mrows / 128), 256, 0, stream>>>(xb, qkvwb,
                                                                    qkv32);
  // 2) l2-normalize q,k (full C), scale + fold 1/temperature into q,
  //    split heads, V transposed
  norm_split_kernel<<<dim3(Mrows), 256, 0, stream>>>(qkv32, temperature, qb, kb,
                                                     vt);
  // 3) flash attention
  attn_kernel<<<dim3(Nc / (16 * 4), Bc * Hc), 128, 0, stream>>>(qb, kb, vt, ob);
  // 4) projection + bias
  gemm_proj_kernel<<<dim3(Cc / 64, Mrows / 128), 256, 0, stream>>>(ob, pwb,
                                                                   proj_b, out);
}
